// MLPDecoder_9526237462975
// MI455X (gfx1250) — matrix operand order repaired and checked by reading, not compiled, not run on hardware
//
#include <hip/hip_runtime.h>

// MLP edge decoder for MI455X (gfx1250, wave32, WMMA f16->f32).
// Each wave processes 32 edges (two 16-row WMMA tiles) per iteration so every
// LDS-resident weight B-fragment is reused for two WMMAs (halves LDS traffic,
// the WGP-level bottleneck).
// Layer1: [32x256]@[256x128]+b1, relu  (128 WMMA / 32 edges)
// Layer2: [32x128]@[128x64]+b2, relu   ( 32 WMMA / 32 edges)
// Layer3: [32x64]@[64x1]+b3            (VALU dot, all 32 lanes)
//
// Fast path (if ws_size allows): z pre-converted to f16 in workspace -> pure
// f16 b128 gathers, no cvt in the hot loop. Fallback converts f32->f16 inline.

typedef __attribute__((ext_vector_type(16))) _Float16 v16h;
typedef __attribute__((ext_vector_type(8)))  float    v8f;

#define NNODES 100000
#define NEDGES 1000000
#define DIM    128     // D
#define K1     256     // 2*D
#define D2     64      // D/2

union Vh {
  v16h v;
  _Float16 h[16];
  unsigned u[8];
};

// ---- prep: z (f32) -> zh (f16), vectorized x4 -------------------------------
__global__ __launch_bounds__(256) void cvt_z_f16(const float* __restrict__ z,
                                                 _Float16* __restrict__ zh,
                                                 int n4) {
  const int i = blockIdx.x * blockDim.x + threadIdx.x;
  if (i < n4) {
    const float4 f = ((const float4*)z)[i];
    union { unsigned long long u; _Float16 h[4]; } p;
    p.h[0] = (_Float16)f.x; p.h[1] = (_Float16)f.y;
    p.h[2] = (_Float16)f.z; p.h[3] = (_Float16)f.w;
    ((unsigned long long*)zh)[i] = p.u;
  }
}

// A-fragment loader: 16x256 f16 row (K<128 from z[src], K>=128 from z[dst]),
// eight 16x32 K-steps in the documented 16-bit A VGPR layout
// (lane group g: VGPR j<4 -> K = koff+8g+2j{,+1}; j>=4 -> K = koff+16+8g+2(j-4){,+1}).
template <bool ZH16>
__device__ __forceinline__ void load_a_frags(v16h (&a)[8],
                                             const float* __restrict__ z,
                                             const _Float16* __restrict__ zh,
                                             int src, int dst, int g) {
  if (ZH16) {
    const _Float16* zs = zh + (size_t)src * DIM;
    const _Float16* zd = zh + (size_t)dst * DIM;
#pragma unroll
    for (int kk = 0; kk < 8; ++kk) {
      const _Float16* base = (kk < 4) ? zs : zd;
      const int koff = (kk & 3) * 32 + 8 * g;
      Vh av;
      const uint4 lo = *(const uint4*)(base + koff);       // halves koff..+7
      const uint4 hi = *(const uint4*)(base + koff + 16);  // halves +16..+23
      av.u[0] = lo.x; av.u[1] = lo.y; av.u[2] = lo.z; av.u[3] = lo.w;
      av.u[4] = hi.x; av.u[5] = hi.y; av.u[6] = hi.z; av.u[7] = hi.w;
      a[kk] = av.v;
    }
  } else {
    const float* zs = z + (size_t)src * DIM;
    const float* zd = z + (size_t)dst * DIM;
#pragma unroll
    for (int kk = 0; kk < 8; ++kk) {
      const float* base = (kk < 4) ? zs : zd;
      const int koff = (kk & 3) * 32 + 8 * g;
      Vh av;
#pragma unroll
      for (int j = 0; j < 8; ++j) {
        const int k2 = koff + ((j < 4) ? (2 * j) : (16 + 2 * (j - 4)));
        const float2 f = *(const float2*)(base + k2);
        av.h[2 * j]     = (_Float16)f.x;
        av.h[2 * j + 1] = (_Float16)f.y;
      }
      a[kk] = av.v;
    }
  }
}

// ---- main kernel ------------------------------------------------------------
template <bool ZH16>
__global__ __launch_bounds__(256) void mlp_edge_decoder(
    const float*    __restrict__ z,
    const _Float16* __restrict__ zh,
    const int*      __restrict__ eidx,   // [2, E]
    const float* __restrict__ W1, const float* __restrict__ b1,
    const float* __restrict__ W2, const float* __restrict__ b2,
    const float* __restrict__ W3, const float* __restrict__ b3,
    float* __restrict__ out)
{
  // Weights staged once per block in LDS (f16), reused across grid-stride loop.
  __shared__ _Float16 w1s[K1 * DIM];     // 64 KB
  __shared__ _Float16 w2s[DIM * D2];     // 16 KB
  __shared__ float b1s[DIM];
  __shared__ float b2s[D2];
  __shared__ float w3s[D2];
  // Per-wave activation scratch: two tiles of 16x128 f16 / 16x64 f32 (4KB each).
  __shared__ __align__(16) unsigned char scratch[8][2 * 4096];  // 8 x 8KB

  const int tid = threadIdx.x;
  for (int i = tid; i < K1 * DIM; i += 256) w1s[i] = (_Float16)W1[i];
  for (int i = tid; i < DIM * D2; i += 256) w2s[i] = (_Float16)W2[i];
  if (tid < DIM) b1s[tid] = b1[tid];
  if (tid < D2)  { b2s[tid] = b2[tid]; w3s[tid] = W3[tid]; }
  __syncthreads();

  const int lane = tid & 31;
  const int wave = tid >> 5;
  const int g    = lane >> 4;   // lane group: K sub-block in A layout / D row group
  const int r    = lane & 15;   // matrix row (M) / column (N) index
  const float bias3 = b3[0];

  _Float16* l1t0 = (_Float16*)(scratch[wave]);
  _Float16* l1t1 = (_Float16*)(scratch[wave] + 4096);
  float*    l2t0 = (float*)(scratch[wave]);
  float*    l2t1 = (float*)(scratch[wave] + 4096);

  const int npairs = (NEDGES + 31) / 32;
  for (int p = blockIdx.x * 8 + wave; p < npairs; p += gridDim.x * 8) {
    // Two 16-edge tiles; clamp tails so EXEC stays all-1 for WMMA.
    int e0 = p * 32 + r;       if (e0 >= NEDGES) e0 = NEDGES - 1;
    int e1 = p * 32 + 16 + r;  if (e1 >= NEDGES) e1 = NEDGES - 1;
    const int src0 = eidx[e0], dst0 = eidx[NEDGES + e0];
    const int src1 = eidx[e1], dst1 = eidx[NEDGES + e1];

    v16h a0[8], a1[8];
    load_a_frags<ZH16>(a0, z, zh, src0, dst0, g);
    load_a_frags<ZH16>(a1, z, zh, src1, dst1, g);

    // ---- Layer 1: each B fragment feeds two WMMAs (tile0 + tile1).
    for (int n = 0; n < 8; ++n) {
      const float bv = b1s[n * 16 + r];
      v8f acc0 = {bv, bv, bv, bv, bv, bv, bv, bv};
      v8f acc1 = acc0;
#pragma unroll
      for (int kk = 0; kk < 8; ++kk) {
        // B layout: lane column N = n*16+r, element s -> K = kk*32 + (s<8 ? 8g+s : 16+8g+s-8).
        v16h bf;
#pragma unroll
        for (int s = 0; s < 16; ++s)
          bf[s] = w1s[(kk * 32 + ((s < 8) ? (8 * g + s) : (16 + 8 * g + (s - 8)))) * DIM + n * 16 + r];
        acc0 = __builtin_amdgcn_wmma_f32_16x16x32_f16(false, a0[kk], false, bf,
                                                      (short)0, acc0, false, false);
        acc1 = __builtin_amdgcn_wmma_f32_16x16x32_f16(false, a1[kk], false, bf,
                                                      (short)0, acc1, false, false);
      }
#pragma unroll
      for (int i = 0; i < 8; ++i) {
        // D layout: VGPR i -> row (i + 8g), column = n*16 + r.
        l1t0[(i + 8 * g) * DIM + n * 16 + r] = (_Float16)fmaxf(acc0[i], 0.0f);
        l1t1[(i + 8 * g) * DIM + n * 16 + r] = (_Float16)fmaxf(acc1[i], 0.0f);
      }
    }

    // ---- Re-fragment layer-1 outputs as A2 (16x128, four 16x32 K-steps each).
    v16h a20[4], a21[4];
#pragma unroll
    for (int kk = 0; kk < 4; ++kk) {
      const int koff = kk * 32 + 8 * g;
      Vh av0, av1;
#pragma unroll
      for (int j = 0; j < 8; ++j) {
        const int k2 = koff + ((j < 4) ? (2 * j) : (16 + 2 * (j - 4)));
        av0.u[j] = *(const unsigned*)&l1t0[r * DIM + k2];
        av1.u[j] = *(const unsigned*)&l1t1[r * DIM + k2];
      }
      a20[kk] = av0.v;
      a21[kk] = av1.v;
    }

    // ---- Layer 2: shared B fragments, store f32 (reuses same LDS scratch;
    // safe: a2 reads issued before these stores, DS is in-order per wave).
    for (int n = 0; n < 4; ++n) {
      const float bv = b2s[n * 16 + r];
      v8f acc0 = {bv, bv, bv, bv, bv, bv, bv, bv};
      v8f acc1 = acc0;
#pragma unroll
      for (int kk = 0; kk < 4; ++kk) {
        v16h bf;
#pragma unroll
        for (int s = 0; s < 16; ++s)
          bf[s] = w2s[(kk * 32 + ((s < 8) ? (8 * g + s) : (16 + 8 * g + (s - 8)))) * D2 + n * 16 + r];
        acc0 = __builtin_amdgcn_wmma_f32_16x16x32_f16(false, a20[kk], false, bf,
                                                      (short)0, acc0, false, false);
        acc1 = __builtin_amdgcn_wmma_f32_16x16x32_f16(false, a21[kk], false, bf,
                                                      (short)0, acc1, false, false);
      }
#pragma unroll
      for (int i = 0; i < 8; ++i) {
        l2t0[(i + 8 * g) * D2 + n * 16 + r] = fmaxf(acc0[i], 0.0f);
        l2t1[(i + 8 * g) * D2 + n * 16 + r] = fmaxf(acc1[i], 0.0f);
      }
    }

    // ---- Layer 3: 64 -> 1 dot; all 32 lanes active (g selects the tile).
    {
      const float* l2 = g ? l2t1 : l2t0;
      float sum = bias3;
#pragma unroll 8
      for (int c = 0; c < D2; ++c) sum += l2[r * D2 + c] * w3s[c];
      const int eo = p * 32 + g * 16 + r;
      if (eo < NEDGES) out[eo] = sum;
    }
  }
}

extern "C" void kernel_launch(void* const* d_in, const int* in_sizes, int n_in,
                              void* d_out, int out_size, void* d_ws, size_t ws_size,
                              hipStream_t stream) {
  (void)in_sizes; (void)n_in; (void)out_size;
  const float* z   = (const float*)d_in[0];
  const int*   ei  = (const int*)d_in[1];
  const float* W1  = (const float*)d_in[2];
  const float* b1  = (const float*)d_in[3];
  const float* W2  = (const float*)d_in[4];
  const float* b2  = (const float*)d_in[5];
  const float* W3  = (const float*)d_in[6];
  const float* b3  = (const float*)d_in[7];
  float* out = (float*)d_out;

  const dim3 grid(2048), block(256);
  const size_t zh_bytes = (size_t)NNODES * DIM * sizeof(_Float16);  // 25.6 MB

  if (d_ws != nullptr && ws_size >= zh_bytes) {
    // Fast path: pre-convert z to f16 in workspace, then WMMA kernel reads f16.
    _Float16* zh = (_Float16*)d_ws;
    const int n4 = NNODES * DIM / 4;
    hipLaunchKernelGGL(cvt_z_f16, dim3((n4 + 255) / 256), block, 0, stream,
                       z, zh, n4);
    hipLaunchKernelGGL((mlp_edge_decoder<true>), grid, block, 0, stream,
                       z, zh, ei, W1, b1, W2, b2, W3, b3, out);
  } else {
    hipLaunchKernelGGL((mlp_edge_decoder<false>), grid, block, 0, stream,
                       z, (const _Float16*)nullptr, ei, W1, b1, W2, b2, W3, b3, out);
  }
}
